// AutoEncoder_17695265259991
// MI455X (gfx1250) — compile-verified, matrix operand order checked by reading, not run on hardware
//
#include <hip/hip_runtime.h>
#include <hip/hip_bf16.h>
#include <stddef.h>
#include <stdint.h>

// ---------------------------------------------------------------------------
// AutoEncoder set-transformer-ish kernel for MI455X (gfx1250, wave32, WMMA).
// All five GEMMs run as V_WMMA_F32_16X16X32_F16 tiles (f16 in, f32 accum).
// ---------------------------------------------------------------------------

#define BSETS 4096
#define MAXN  64
#define DIMF  256
#define WD    32

typedef __attribute__((ext_vector_type(16))) _Float16 v16h;
typedef __attribute__((ext_vector_type(8)))  float    v8f;

// LDS row strides (halves), padded to dodge bank conflicts
#define XSTR 264   // Xpos rows (256 + 8)
#define YSTR 36    // Yk/Yv/Z/decoded rows (32 + 4)

// Workspace byte offsets (all 256-aligned)
#define OFF_PTR    0          // (BSETS+1) * int
#define OFF_PE     16640      // 64*256 float
#define OFF_QF32   82176      // 64*32 float
#define OFF_FRAGQ  90368      // 4 A-frags  * 512 halves
#define OFF_FRAGWK 94464      // 16 B-frags * 512 halves
#define OFF_FRAGWV 110848     // 16 B-frags * 512 halves
#define OFF_FRAGWM 127232     // 16 B-frags * 512 halves

// ISA 7.12.2: 16-bit A-matrix 16x32 per-lane K index for half h (0..15).
// lane<16 holds K {0..7,16..23}; lane>=16 holds K {8..15,24..31}.
// B (32x16) assumed symmetric: N = lane&15, same K striping.
__device__ __forceinline__ int a_k_of(int lane, int h) {
    int p = h >> 1, lo = h & 1, kh = lane >> 4;
    int base = (p < 4) ? (p * 2) : (16 + (p - 4) * 2);
    return base + kh * 8 + lo;
}

__device__ __forceinline__ v16h wmma_ldA(const _Float16* X, int stride, int mt, int kt, int lane) {
    v16h a;
    const _Float16* p = X + (mt * 16 + (lane & 15)) * stride + kt * 32;
#pragma unroll
    for (int h = 0; h < 16; ++h) a[h] = p[a_k_of(lane, h)];
    return a;
}

// A-fragment of Y^T where Y is stored [rows=K][cols=M] in LDS
__device__ __forceinline__ v16h wmma_ldA_T(const _Float16* Y, int stride, int mt, int kt, int lane) {
    v16h a;
    int i = mt * 16 + (lane & 15);
#pragma unroll
    for (int h = 0; h < 16; ++h) a[h] = Y[(kt * 32 + a_k_of(lane, h)) * stride + i];
    return a;
}

__device__ __forceinline__ v16h wmma_ldB_lds(const _Float16* Bm, int stride, int kt, int nt, int lane) {
    v16h b;
    int nn = nt * 16 + (lane & 15);
#pragma unroll
    for (int h = 0; h < 16; ++h) b[h] = Bm[(kt * 32 + a_k_of(lane, h)) * stride + nn];
    return b;
}

// Pre-swizzled fragment in global memory: (frag*32 + lane) * 16 halves, 32B/lane
__device__ __forceinline__ v16h gfrag(const _Float16* p0, int fragIdx, int lane) {
    return reinterpret_cast<const v16h*>(p0)[fragIdx * 32 + lane];
}

// C/D 16x16 f32 layout: VGPR r -> M = r + (lane>=16 ? 8 : 0), N = lane&15
__device__ __forceinline__ void storeC_lds(_Float16* Y, int stride, int mt, int nt, int lane, v8f c) {
    int nn = nt * 16 + (lane & 15);
    int mb = mt * 16 + ((lane >> 4) << 3);
#pragma unroll
    for (int r = 0; r < 8; ++r) Y[(mb + r) * stride + nn] = (_Float16)c[r];
}

__device__ __forceinline__ v8f wmma16(v16h a, v16h b, v8f c) {
    return __builtin_amdgcn_wmma_f32_16x16x32_f16(false, a, false, b, (short)0, c, false, false);
}

// ------------------------------ setup kernels ------------------------------

__global__ void k_ptr(const int* __restrict__ batch, int N, int* __restrict__ ptr) {
    int t = blockIdx.x * blockDim.x + threadIdx.x;
    if (t > BSETS) return;
    int lo = 0, hi = N;                 // lower_bound(batch, t); batch sorted
    while (lo < hi) { int mid = (lo + hi) >> 1; if (batch[mid] < t) lo = mid + 1; else hi = mid; }
    ptr[t] = lo;
}

__global__ void k_pe(float* __restrict__ pe) {
    int idx = blockIdx.x * blockDim.x + threadIdx.x;   // 64*256
    if (idx >= MAXN * DIMF) return;
    int p = idx >> 8, i = idx & 255;
    float angle = (float)p * __powf(10000.0f, -(float)(2 * (i >> 1)) / (float)DIMF);
    pe[idx] = (i & 1) ? __cosf(angle) : __sinf(angle);
}

__global__ void k_query(const float* __restrict__ pe, const float* __restrict__ Wq,
                        float* __restrict__ q) {
    int idx = blockIdx.x * blockDim.x + threadIdx.x;   // 64*32
    if (idx >= MAXN * WD) return;
    int m = idx >> 5, k = idx & 31;
    float s = 0.f;
    for (int d = 0; d < DIMF; ++d) s += pe[m * DIMF + d] * Wq[k * DIMF + d];
    q[idx] = s;
}

__global__ void k_fragA_query(const float* __restrict__ q, _Float16* __restrict__ dst) {
    int idx = blockIdx.x * blockDim.x + threadIdx.x;   // 4 frags * 512
    if (idx >= 4 * 512) return;
    int h = idx & 15, lane = (idx >> 4) & 31, mt = idx >> 9;
    dst[idx] = (_Float16)q[(mt * 16 + (lane & 15)) * WD + a_k_of(lane, h)];
}

// B-fragments for yk/yv projection: B[k=d][n=w] = W[w][d]; frag = kt*2+nt (8x2)
__global__ void k_fragB_proj(const float* __restrict__ Wm, _Float16* __restrict__ dst) {
    int idx = blockIdx.x * blockDim.x + threadIdx.x;   // 16 frags * 512
    if (idx >= 16 * 512) return;
    int h = idx & 15, lane = (idx >> 4) & 31, f = idx >> 9;
    int w = (f & 1) * 16 + (lane & 15);
    int d = (f >> 1) * 32 + a_k_of(lane, h);
    dst[idx] = (_Float16)Wm[w * DIMF + d];
}

// B-fragments for output map: B[k=w][n=d] = Wmap[d][w]; frag = dt (1x16)
__global__ void k_fragB_map(const float* __restrict__ Wmap, _Float16* __restrict__ dst) {
    int idx = blockIdx.x * blockDim.x + threadIdx.x;   // 16 frags * 512
    if (idx >= 16 * 512) return;
    int h = idx & 15, lane = (idx >> 4) & 31, f = idx >> 9;
    int d = f * 16 + (lane & 15);
    int w = a_k_of(lane, h);
    dst[idx] = (_Float16)Wmap[d * WD + w];
}

__global__ void k_tail_batch(const int* __restrict__ batch, float* __restrict__ dst, int N) {
    int i = blockIdx.x * blockDim.x + threadIdx.x;
    if (i < N) dst[i] = (float)batch[i];
}

// ------------------------------- main kernel -------------------------------

__launch_bounds__(256)
__global__ void k_main(const float* __restrict__ x, const int* __restrict__ ptr,
                       const float* __restrict__ pe,
                       const _Float16* __restrict__ fragQ, const _Float16* __restrict__ fragWk,
                       const _Float16* __restrict__ fragWv, const _Float16* __restrict__ fragWm,
                       const float* __restrict__ bmap, float* __restrict__ out) {
    __shared__ _Float16 Xs[MAXN * XSTR];   // Xpos f16 (padded rows zero = mask)
    __shared__ _Float16 Yk[MAXN * YSTR];
    __shared__ _Float16 Yv[MAXN * YSTR];
    __shared__ _Float16 Zs[WD * YSTR];
    __shared__ _Float16 Dc[MAXN * YSTR];

    const int b    = blockIdx.x;
    const int base = ptr[b];
    const int n    = ptr[b + 1] - base;
    const int tid  = threadIdx.x;
    const int lane = tid & 31;            // wave32
    const int wave = tid >> 5;            // 8 waves / block

    // Phase 1: Xpos = f16(x + PE) for valid rows, zero for padded rows.
    // 4 halves packed per 8-byte LDS store (row stride 528 B, c4*2 B: both 8B-aligned).
    for (int q = tid; q < MAXN * (DIMF / 4); q += 256) {
        int row = q >> 6, c4 = (q & 63) * 4;
        union { _Float16 h[4]; uint2 u; } pk;
        if (row < n) {
            float4 xv = *reinterpret_cast<const float4*>(x + (size_t)(base + row) * DIMF + c4);
            float4 pv = *reinterpret_cast<const float4*>(pe + row * DIMF + c4);
            pk.h[0] = (_Float16)(xv.x + pv.x);
            pk.h[1] = (_Float16)(xv.y + pv.y);
            pk.h[2] = (_Float16)(xv.z + pv.z);
            pk.h[3] = (_Float16)(xv.w + pv.w);
        } else {
            pk.u = make_uint2(0u, 0u);
        }
        *reinterpret_cast<uint2*>(Xs + row * XSTR + c4) = pk.u;
    }
    __syncthreads();

    // Phase 2: Yk = Xpos·Wkᵀ, Yv = Xpos·Wvᵀ  [64,256]x[256,32]
    for (int j = wave; j < 16; j += 8) {
        const _Float16* fw = (j & 8) ? fragWv : fragWk;
        _Float16*       Yd = (j & 8) ? Yv : Yk;
        int t = j & 7, mt = t >> 1, nt = t & 1;
        v8f c = {};
#pragma unroll
        for (int kt = 0; kt < 8; ++kt) {
            v16h a  = wmma_ldA(Xs, XSTR, mt, kt, lane);
            v16h bb = gfrag(fw, kt * 2 + nt, lane);
            c = wmma16(a, bb, c);
        }
        storeC_lds(Yd, YSTR, mt, nt, lane, c);
    }
    __syncthreads();

    // Phase 3: Z = Yvᵀ·Yk  [32,64]x[64,32]
    if (wave < 4) {
        int mi = wave >> 1, nj = wave & 1;
        v8f c = {};
#pragma unroll
        for (int kt = 0; kt < 2; ++kt) {
            v16h a  = wmma_ldA_T(Yv, YSTR, mi, kt, lane);
            v16h bb = wmma_ldB_lds(Yk, YSTR, kt, nj, lane);
            c = wmma16(a, bb, c);
        }
        storeC_lds(Zs, YSTR, mi, nj, lane, c);
    }
    __syncthreads();

    // Phase 4: decoded = query·Z  [64,32]x[32,32]
    {
        int mt = wave >> 1, nt = wave & 1;
        v16h a  = gfrag(fragQ, mt, lane);
        v16h bb = wmma_ldB_lds(Zs, YSTR, 0, nt, lane);
        v8f  c  = {};
        c = wmma16(a, bb, c);
        storeC_lds(Dc, YSTR, mt, nt, lane, c);
    }
    __syncthreads();

    // Phase 5: out = decoded·Wmapᵀ + bmap, write valid rows only
    for (int t = wave; t < 64; t += 8) {
        int mt = t >> 4, dt = t & 15;
        v16h a  = wmma_ldA(Dc, YSTR, mt, 0, lane);
        v16h bb = gfrag(fragWm, dt, lane);
        v8f  c  = {};
        c = wmma16(a, bb, c);
        int col = dt * 16 + (lane & 15);
        float bm = bmap[col];
        int mb = mt * 16 + ((lane >> 4) << 3);
#pragma unroll
        for (int r = 0; r < 8; ++r) {
            int row = mb + r;
            if (row < n) out[(size_t)(base + row) * DIMF + col] = c[r] + bm;
        }
    }
}

// ------------------------------ launch glue --------------------------------

extern "C" void kernel_launch(void* const* d_in, const int* in_sizes, int n_in,
                              void* d_out, int out_size, void* d_ws, size_t ws_size,
                              hipStream_t stream) {
    const float* x     = (const float*)d_in[0];
    const int*   batch = (const int*)  d_in[1];
    const float* Wk    = (const float*)d_in[2];
    const float* Wv    = (const float*)d_in[3];
    const float* Wq    = (const float*)d_in[4];
    const float* Wmap  = (const float*)d_in[5];
    const float* bmap  = (const float*)d_in[6];
    float*       out   = (float*)d_out;
    const int N = in_sizes[1];

    char* ws = (char*)d_ws;
    int*      ptr    = (int*)     (ws + OFF_PTR);
    float*    pe     = (float*)   (ws + OFF_PE);
    float*    qf32   = (float*)   (ws + OFF_QF32);
    _Float16* fragQ  = (_Float16*)(ws + OFF_FRAGQ);
    _Float16* fragWk = (_Float16*)(ws + OFF_FRAGWK);
    _Float16* fragWv = (_Float16*)(ws + OFF_FRAGWV);
    _Float16* fragWm = (_Float16*)(ws + OFF_FRAGWM);

    k_ptr        <<<(BSETS + 1 + 255) / 256, 256, 0, stream>>>(batch, N, ptr);
    k_pe         <<<(MAXN * DIMF + 255) / 256, 256, 0, stream>>>(pe);
    k_query      <<<(MAXN * WD + 255) / 256, 256, 0, stream>>>(pe, Wq, qf32);
    k_fragA_query<<<(4 * 512 + 255) / 256, 256, 0, stream>>>(qf32, fragQ);
    k_fragB_proj <<<(16 * 512 + 255) / 256, 256, 0, stream>>>(Wk, fragWk);
    k_fragB_proj <<<(16 * 512 + 255) / 256, 256, 0, stream>>>(Wv, fragWv);
    k_fragB_map  <<<(16 * 512 + 255) / 256, 256, 0, stream>>>(Wmap, fragWm);

    k_main<<<BSETS, 256, 0, stream>>>(x, ptr, pe, fragQ, fragWk, fragWv, fragWm, bmap, out);

    // Second tuple output (batch), if the harness allotted space for it.
    if (out_size >= N * DIMF + N)
        k_tail_batch<<<(N + 255) / 256, 256, 0, stream>>>(batch, out + (size_t)N * DIMF, N);
}
